// Encoder_81097572483396
// MI455X (gfx1250) — matrix operand order repaired and checked by reading, not compiled, not run on hardware
//
#include <hip/hip_runtime.h>
#include <hip/hip_bf16.h>

// Problem constants (from reference): B=2048, T=128, D=20, H=128
constexpr int kB  = 2048;
constexpr int kT  = 128;
constexpr int kD  = 20;
constexpr int kH  = 128;
constexpr int kG4 = 4 * kH;        // 512 gate columns
constexpr int kKC = 160;           // K = D + H = 148, padded to 5*32
constexpr int kBM = 16;            // batch rows per workgroup (1 M-tile)

typedef __attribute__((ext_vector_type(16))) __bf16 v16bf;
typedef __attribute__((ext_vector_type(8)))  __bf16 v8bf;
typedef __attribute__((ext_vector_type(8)))  float  v8f;

__device__ __forceinline__ float fast_rcp(float x) {
  return __builtin_amdgcn_rcpf(x);
}
__device__ __forceinline__ float fast_sigmoid(float x) {
  return fast_rcp(1.f + __expf(-x));          // v_exp_f32 + v_rcp_f32, no branches
}
__device__ __forceinline__ float fast_tanh(float x) {
  return 1.f - 2.f * fast_rcp(__expf(2.f * x) + 1.f);
}

// ---------------------------------------------------------------------------
// Prep 1: build K-major bf16 fused weight Wt[k][n] = [W_ih | W_hh]^T (padded),
//         and fused bias btot = b_ih + b_hh.
// ---------------------------------------------------------------------------
__global__ void prep_wt_bias(const float* __restrict__ W_ih,
                             const float* __restrict__ W_hh,
                             const float* __restrict__ b_ih,
                             const float* __restrict__ b_hh,
                             __bf16* __restrict__ Wt,
                             float* __restrict__ btot) {
  int idx = blockIdx.x * blockDim.x + threadIdx.x;
  if (idx < kKC * kG4) {
    int k = idx / kG4, n = idx % kG4;
    float v = 0.f;
    if (k < kD)            v = W_ih[n * kD + k];
    else if (k < kD + kH)  v = W_hh[n * kH + (k - kD)];
    Wt[idx] = (__bf16)v;
  }
  if (idx < kG4) btot[idx] = b_ih[idx] + b_hh[idx];
}

// ---------------------------------------------------------------------------
// Prep 2: xw[b,d] = sum_t X[b,t,d] * w_x[t]   (w_x = W_attn[2H:2H+T])
// ---------------------------------------------------------------------------
__global__ void prep_xw(const float* __restrict__ X,
                        const float* __restrict__ W_attn,
                        float* __restrict__ xw) {
  int idx = blockIdx.x * blockDim.x + threadIdx.x;
  if (idx >= kB * kD) return;
  int b = idx / kD, d = idx % kD;
  float s = 0.f;
  for (int t = 0; t < kT; ++t)
    s += X[(b * kT + t) * kD + d] * W_attn[2 * kH + t];
  xw[idx] = s;
}

// ---------------------------------------------------------------------------
// Main: recurrent attention-LSTM. 128 workgroups x 256 threads (8 waves).
// Wave w owns gate tiles {g*8+w : g=0..3}; weight B-fragments persist in VGPRs
// across all 127 timesteps; c state lives in D-fragment-layout registers.
// A tile is double-buffered so h(t) stores never conflict with step-t reads:
// only two barriers per timestep.
// ---------------------------------------------------------------------------
__global__ __launch_bounds__(256) void lstm_attn(
    const float* __restrict__ X,
    const float* __restrict__ W_attn,
    const float* __restrict__ b_attn,
    const __bf16* __restrict__ Wt,     // (kKC x kG4) K-major bf16
    const float* __restrict__ xw,      // (kB x kD)
    const float* __restrict__ btot,    // (kG4)
    float* __restrict__ out_xt,        // (kB, kT-1, kD)
    float* __restrict__ out_xe)        // (kB, kT-1, kH)
{
  __shared__ __align__(16) __bf16 A[2][kBM][kKC]; // ping-pong [x_tilde|h|pad]
  __shared__ float hs_part[8][kBM];               // per-wave hs partials

  const int tid  = threadIdx.x;
  const int lane = tid & 31;
  const int w    = tid >> 5;           // wave id 0..7
  const int half = lane >> 4;          // 0 | 1 (wave32 halves)
  const int l16  = lane & 15;
  const int b0   = blockIdx.x * kBM;
  const int col  = w * 16 + l16;       // owned h/c column 0..127

  // ---- one-time: load B-matrix fragments (4 gate tiles x 5 K-blocks) ------
  // B 32x16 layout: element e -> K = kb*32 + 16*(e/8) + 8*half + e%8, N = tile*16 + l16
  v16bf bfrag[4][5];
  for (int g = 0; g < 4; ++g) {
    const int N = g * kH + col;
    for (int kb = 0; kb < 5; ++kb) {
      const int kbase = kb * 32 + 8 * half;
      v16bf f;
      #pragma unroll
      for (int e = 0; e < 16; ++e) f[e] = Wt[(kbase + (e < 8 ? e : e + 8)) * kG4 + N];
      bfrag[g][kb] = f;
    }
  }

  float bias[4];
  #pragma unroll
  for (int g = 0; g < 4; ++g) bias[g] = btot[g * kH + col];
  const float wh_l  = W_attn[col];          // w_h[col]
  const float ws_l  = W_attn[kH + col];     // w_s[col]
  const float battn = b_attn[0];

  float xwr[kD];
  if (tid < kBM) {
    #pragma unroll
    for (int d = 0; d < kD; ++d) xwr[d] = xw[(b0 + tid) * kD + d];
  }
  for (int i = tid; i < 2 * kBM * kKC; i += 256) (&A[0][0][0])[i] = (__bf16)0.f;
  if (tid < 8 * kBM) hs_part[tid >> 4][tid & 15] = 0.f;

  v8f c;                                     // cell state, D-fragment layout
  #pragma unroll
  for (int r = 0; r < 8; ++r) c[r] = 0.f;

  __syncthreads();

  for (int t = 0; t < kT - 1; ++t) {
    const int cur = t & 1;
    const int nxt = cur ^ 1;

    // -- phase 1: softmax attention + x_tilde (one batch row per thread) ----
    if (tid < kBM) {
      const int b = b0 + tid;
      float hsr = battn;                 // hs = b_attn + h.w_h + c.w_s
      #pragma unroll
      for (int ww = 0; ww < 8; ++ww) hsr += hs_part[ww][tid];
      float xv[kD], ev[kD], m = -1e30f;
      #pragma unroll
      for (int d = 0; d < kD; ++d) {
        xv[d] = X[(b * kT + t) * kD + d];
        ev[d] = hsr + xwr[d];
        m = fmaxf(m, ev[d]);
      }
      float s = 0.f;
      #pragma unroll
      for (int d = 0; d < kD; ++d) { ev[d] = __expf(ev[d] - m); s += ev[d]; }
      const float inv = fast_rcp(s);
      #pragma unroll
      for (int d = 0; d < kD; ++d) {
        float xt = ev[d] * inv * xv[d];
        A[cur][tid][d] = (__bf16)xt;
        out_xt[(b * (kT - 1) + t) * kD + d] = xt;
      }
      if (t + 1 < kT - 1) __builtin_prefetch(&X[(b * kT + t + 1) * kD], 0, 0);
    }
    __syncthreads();   // x_tilde(t) + h(t-1) + hs_part(t-1) visible to all

    // -- phase 2: A fragments from LDS (two ds_load_b128 per K-block) -------
    // A 16x32 layout: element e -> K = kb*32 + 16*(e/8) + 8*half + e%8
    v16bf afrag[5];
    #pragma unroll
    for (int kb = 0; kb < 5; ++kb) {
      const int kbase = kb * 32 + 8 * half;
      v8bf lo = *(const v8bf*)&A[cur][l16][kbase];
      v8bf hi = *(const v8bf*)&A[cur][l16][kbase + 16];
      v16bf a;
      #pragma unroll
      for (int e = 0; e < 8; ++e) { a[e] = lo[e]; a[8 + e] = hi[e]; }
      afrag[kb] = a;
    }

    // -- phase 3: gates = [x_tilde|h] @ Wt  (20 WMMA per wave per step) -----
    v8f acc[4];
    #pragma unroll
    for (int g = 0; g < 4; ++g)
      #pragma unroll
      for (int r = 0; r < 8; ++r) acc[g][r] = bias[g];
    #pragma unroll
    for (int kb = 0; kb < 5; ++kb)
      #pragma unroll
      for (int g = 0; g < 4; ++g)
        acc[g] = __builtin_amdgcn_wmma_f32_16x16x32_bf16(
            false, afrag[kb], false, bfrag[g][kb],
            (short)0, acc[g], false, false);

    // -- phase 4: LSTM cell, branch-free, in registers ----------------------
    float p[8];
    v8f hnew;
    #pragma unroll
    for (int r = 0; r < 8; ++r) {
      float ig = fast_sigmoid(acc[0][r]);
      float fg = fast_sigmoid(acc[1][r]);
      float gg = fast_tanh(acc[2][r]);
      float og = fast_sigmoid(acc[3][r]);
      float cn = fg * c[r] + ig * gg;
      float hn = og * fast_tanh(cn);
      c[r]    = cn;
      hnew[r] = hn;
      p[r] = hn * wh_l + cn * ws_l;  // partial of next hs
    }
    // reduce hs partials across the 16 columns held by each wave half
    #pragma unroll
    for (int r = 0; r < 8; ++r)
      #pragma unroll
      for (int msk = 1; msk < 16; msk <<= 1)
        p[r] += __shfl_xor(p[r], msk, 32);

    // -- phase 5: publish h into NEXT buffer, hs partials, global stores ----
    #pragma unroll
    for (int r = 0; r < 8; ++r) {
      const int M = r + 8 * half;
      A[nxt][M][kD + col] = (__bf16)hnew[r];
      out_xe[((b0 + M) * (kT - 1) + t) * kH + col] = hnew[r];
    }
    if (l16 == 0) {
      #pragma unroll
      for (int r = 0; r < 8; ++r)
        hs_part[w][r + 8 * half] = p[r];      // plain ds_store, no atomics
    }
    __syncthreads();   // h(t), hs_part(t) complete before next phase 1
  }
}

// ---------------------------------------------------------------------------
extern "C" void kernel_launch(void* const* d_in, const int* in_sizes, int n_in,
                              void* d_out, int out_size, void* d_ws, size_t ws_size,
                              hipStream_t stream) {
  (void)in_sizes; (void)n_in; (void)out_size; (void)ws_size;
  const float* X      = (const float*)d_in[0];
  const float* W_attn = (const float*)d_in[1];
  const float* b_attn = (const float*)d_in[2];
  const float* W_ih   = (const float*)d_in[3];
  const float* W_hh   = (const float*)d_in[4];
  const float* b_ih   = (const float*)d_in[5];
  const float* b_hh   = (const float*)d_in[6];

  char* ws = (char*)d_ws;
  __bf16* Wt   = (__bf16*)ws;                     // 160*512*2 = 163840 B
  float*  xw   = (float*)(ws + 163840);           // 2048*20*4 = 163840 B
  float*  btot = (float*)(ws + 327680);           // 512*4     =   2048 B

  float* out_xt = (float*)d_out;
  float* out_xe = out_xt + (size_t)kB * (kT - 1) * kD;

  prep_wt_bias<<<(kKC * kG4 + 255) / 256, 256, 0, stream>>>(
      W_ih, W_hh, b_ih, b_hh, Wt, btot);
  prep_xw<<<(kB * kD + 255) / 256, 256, 0, stream>>>(X, W_attn, xw);
  lstm_attn<<<kB / kBM, 256, 0, stream>>>(
      X, W_attn, b_attn, Wt, xw, btot, out_xt, out_xe);
}
